// Model_41712722379096
// MI455X (gfx1250) — matrix operand order repaired and checked by reading, not compiled, not run on hardware
//
#include <hip/hip_runtime.h>

// ---------------------------------------------------------------------------
// Model constants (from the reference)
// ---------------------------------------------------------------------------
#define B_     32
#define L_     1024
#define ENC    21
#define TF     4
#define DM_    256
#define DI_    512
#define DST_   64
#define HD_    64
#define NH_    8
#define DC_    4
#define NL_    4
#define PRED_  192
#define CO_    21
#define NCH    16            // L_/64 chunks
#define DIP_   1160          // 2*DI + 2*DST + NH
#define CONVD_ 640           // DI + 2*DST
#define TOK    (B_*L_)       // 32768 tokens

typedef __bf16 bf16;
typedef __attribute__((ext_vector_type(16))) __bf16 v16bf;
typedef __attribute__((ext_vector_type(8)))  __bf16 v8bf;
typedef __attribute__((ext_vector_type(8)))  float  v8f;

__device__ inline v8f vzero8f() { v8f z; for (int i = 0; i < 8; ++i) z[i] = 0.f; return z; }

// ---------------------------------------------------------------------------
// CDNA5 async global->LDS copy (16B per lane), tracked by ASYNCcnt.
// LDS address = low 32 bits of the flat shared pointer (ISA 10.2 aperture map).
// ---------------------------------------------------------------------------
__device__ inline void async_g2l_b128(const void* g, void* l) {
  unsigned lds = (unsigned)(uintptr_t)l;
  asm volatile("global_load_async_to_lds_b128 %0, %1, off"
               :: "v"(lds), "v"(g) : "memory");
}
__device__ inline void wait_async0() {
  asm volatile("s_wait_asynccnt 0x0" ::: "memory");
}

// A-operand fragment (M x K, 16x32 bf16). ISA 7.12.2: lanes 0-15 hold row r,
// K halves {0..7, 16..23}; lanes 16-31 hold row r, K halves {8..15, 24..31}.
__device__ inline v16bf ld_frag_a(const bf16* s, int row0, int k0, int ld) {
  int lane = threadIdx.x & 31;
  int r = lane & 15, hi = lane >> 4;
  const bf16* p = s + (size_t)(row0 + r) * ld + k0 + hi * 8;
  union { v16bf v; v8bf h[2]; } u;
  u.h[0] = *(const v8bf*)p;
  u.h[1] = *(const v8bf*)(p + 16);
  return u.v;
}

// B-operand fragment. Source stored row-major as (N x K) (i.e. rows of B^T).
// Lanes 0-15: col = lane, K {0..7, 16..23}; lanes 16-31: col = lane-16, K {8..15, 24..31}.
__device__ inline v16bf ld_frag_b(const bf16* s, int col0, int k0, int ld) {
  int lane = threadIdx.x & 31;
  int r = lane & 15, hi = lane >> 4;
  const bf16* p = s + (size_t)(col0 + r) * ld + k0 + hi * 8;
  union { v16bf v; v8bf h[2]; } u;
  u.h[0] = *(const v8bf*)p;
  u.h[1] = *(const v8bf*)(p + 16);
  return u.v;
}

__device__ inline v8f wmma_bf16(v16bf a, v16bf b, v8f c) {
  // D = A*B + C, f32 accumulate. 8-arg form: neg_a, A, neg_b, B, c_mod, C, reuse_a, reuse_b
  return __builtin_amdgcn_wmma_f32_16x16x32_bf16(false, a, false, b, (short)0, c, false, false);
}

__device__ inline float siluf(float x) { return x / (1.f + __expf(-x)); }

// ---------------------------------------------------------------------------
// GEMM: C[M,N] = A[M,K] @ W[N,K]^T    (A, W bf16 row-major; C f32)
// Block: 256 threads = 8 waves; block tile 128(M) x 64(N); BK = 32.
// A tile staged with global_load_async_to_lds_b128 (ASYNCcnt path);
// W tile staged via guarded load + ds_store (needs zero fill at N edge).
// ---------------------------------------------------------------------------
__global__ __launch_bounds__(256) void gemm_bf16_nt(
    const bf16* __restrict__ A, const bf16* __restrict__ W,
    float* __restrict__ C, int M, int N, int K) {
  __shared__ alignas(16) bf16 sA[128 * 40];
  __shared__ alignas(16) bf16 sB[64 * 40];
  const int tid  = threadIdx.x;
  const int bm   = blockIdx.x * 128;
  const int bn   = blockIdx.y * 64;
  const int wave = tid >> 5;
  const int wm   = (wave & 3) * 32;
  const int wn   = (wave >> 2) * 32;
  v8f acc00 = vzero8f(), acc01 = vzero8f(), acc10 = vzero8f(), acc11 = vzero8f();
  const int ar = tid >> 1, ac = (tid & 1) * 16;   // A tile: 128 rows x 32 halves
  const int br = tid >> 2, bc = (tid & 3) * 8;    // B tile:  64 rows x 32 halves
  for (int k0 = 0; k0 < K; k0 += 32) {
    __syncthreads();   // previous iteration's fragment reads complete
    {
      const bf16* g = A + (size_t)(bm + ar) * K + k0 + ac;
      async_g2l_b128(g,     &sA[ar * 40 + ac]);
      async_g2l_b128(g + 8, &sA[ar * 40 + ac + 8]);
      if (k0 + 32 < K) __builtin_prefetch(g + 32, 0, 3);   // global_prefetch_b8
    }
    {
      int n = bn + br;
      v8bf val;
      if (n < N) val = *(const v8bf*)(W + (size_t)n * K + k0 + bc);
      else       for (int i = 0; i < 8; ++i) val[i] = (bf16)0.0f;
      *(v8bf*)(&sB[br * 40 + bc]) = val;
    }
    wait_async0();     // ASYNCcnt == 0: async LDS writes landed
    __syncthreads();   // publish tile to all waves
    v16bf a0 = ld_frag_a(sA, wm,      0, 40);
    v16bf a1 = ld_frag_a(sA, wm + 16, 0, 40);
    v16bf b0 = ld_frag_b(sB, wn,      0, 40);
    v16bf b1 = ld_frag_b(sB, wn + 16, 0, 40);
    acc00 = wmma_bf16(a0, b0, acc00);
    acc01 = wmma_bf16(a0, b1, acc01);
    acc10 = wmma_bf16(a1, b0, acc10);
    acc11 = wmma_bf16(a1, b1, acc11);
  }
  const int lane = tid & 31, r = lane & 15, hi = lane >> 4;
  for (int v = 0; v < 8; ++v) {
    int row0 = bm + wm + v + 8 * hi;
    int row1 = row0 + 16;
    int c0 = bn + wn + r, c1 = c0 + 16;
    if (c0 < N) { C[(size_t)row0 * N + c0] = acc00[v]; C[(size_t)row1 * N + c0] = acc10[v]; }
    if (c1 < N) { C[(size_t)row0 * N + c1] = acc01[v]; C[(size_t)row1 * N + c1] = acc11[v]; }
  }
}

// ---------------------------------------------------------------------------
// Per-(batch, channel) instance-norm statistics over L
// ---------------------------------------------------------------------------
__global__ __launch_bounds__(256) void stats_kernel(
    const float* __restrict__ x, float* __restrict__ meanb, float* __restrict__ stdb) {
  __shared__ float r1[256], r2[256];
  int b = blockIdx.x / ENC, ch = blockIdx.x % ENC;
  int tid = threadIdx.x;
  float s1 = 0.f, s2 = 0.f;
  for (int l = tid; l < L_; l += 256) {
    float v = x[((size_t)b * L_ + l) * ENC + ch];
    s1 += v; s2 += v * v;
  }
  r1[tid] = s1; r2[tid] = s2;
  __syncthreads();
  for (int s = 128; s > 0; s >>= 1) {
    if (tid < s) { r1[tid] += r1[tid + s]; r2[tid] += r2[tid + s]; }
    __syncthreads();
  }
  if (tid == 0) {
    float m   = r1[0] / (float)L_;
    float var = r2[0] / (float)L_ - m * m;
    meanb[b * ENC + ch] = m;
    stdb[b * ENC + ch]  = sqrtf(var + 1e-5f);
  }
}

__global__ void cvt_bf16(const float* __restrict__ src, bf16* __restrict__ dst, int n) {
  int i = blockIdx.x * 256 + threadIdx.x;
  if (i < n) dst[i] = (bf16)src[i];
}

// ---------------------------------------------------------------------------
// Embedding: circular 3-tap token conv on normalized input + PE + temporal
// ---------------------------------------------------------------------------
__global__ __launch_bounds__(256) void embed_kernel(
    const float* __restrict__ x, const float* __restrict__ xmark,
    const float* __restrict__ tokw, const float* __restrict__ tembw,
    const float* __restrict__ meanb, const float* __restrict__ stdb,
    float* __restrict__ hf, bf16* __restrict__ hb) {
  __shared__ float sx[3 * ENC];
  int bl = blockIdx.x;
  int b = bl >> 10, l = bl & 1023;
  int tid = threadIdx.x;
  if (tid < 3 * ENC) {
    int k = tid / ENC, c = tid % ENC;
    int lm = (l + k - 1 + L_) % L_;  // wrap pad by 1 each side
    float v = x[((size_t)b * L_ + lm) * ENC + c];
    sx[tid] = (v - meanb[b * ENC + c]) / stdb[b * ENC + c];
  }
  __syncthreads();
  int d = tid;
  float acc = 0.f;
  for (int k = 0; k < 3; ++k)
    for (int c = 0; c < ENC; ++c)
      acc += sx[k * ENC + c] * tokw[d * (ENC * 3) + c * 3 + k];
  int j2 = d & ~1;
  float freq = __expf(-(float)j2 * (9.210340371976184f / (float)DM_)); // ln(10000)
  float ang  = (float)l * freq;
  acc += (d & 1) ? cosf(ang) : sinf(ang);
  for (int t = 0; t < TF; ++t)
    acc += xmark[(size_t)bl * TF + t] * tembw[d * TF + t];
  hf[(size_t)bl * DM_ + d] = acc;
  hb[(size_t)bl * DM_ + d] = (bf16)acc;
}

// ---------------------------------------------------------------------------
// Causal depthwise conv (4 taps) + bias + SiLU on xBC; softplus dt
// ---------------------------------------------------------------------------
__global__ __launch_bounds__(256) void conv_silu_kernel(
    const float* __restrict__ zx, const float* __restrict__ cw,
    const float* __restrict__ cb, const float* __restrict__ dtbias,
    bf16* __restrict__ xbc, float* __restrict__ dtb) {
  int bl = blockIdx.x;
  int b = bl >> 10, l = bl & 1023;
  int tid = threadIdx.x;
  for (int ch = tid; ch < CONVD_; ch += 256) {
    float acc = cb[ch];
    for (int k = 0; k < DC_; ++k) {
      int ls = l - (DC_ - 1) + k;
      if (ls >= 0)
        acc += zx[((size_t)(b * L_ + ls)) * DIP_ + DI_ + ch] * cw[ch * DC_ + k];
    }
    xbc[(size_t)bl * CONVD_ + ch] = (bf16)siluf(acc);
  }
  if (tid < NH_) {
    float raw = zx[(size_t)bl * DIP_ + DI_ + CONVD_ + tid] + dtbias[tid];
    float sp  = raw > 20.f ? raw : log1pf(__expf(raw));
    dtb[(size_t)bl * NH_ + tid] = sp;
  }
}

// ---------------------------------------------------------------------------
// SSD part 1, per (b, chunk):  CB = C @ Bm^T (shared over heads);
// per-head states[p,n] = sum_q Xdt[q,p] * Bm[q,n] * exp(Atot - Acum[q])
// ---------------------------------------------------------------------------
__global__ __launch_bounds__(256) void ssd_part1(
    const bf16* __restrict__ xbc, const float* __restrict__ dtb,
    const float* __restrict__ alog, float* __restrict__ cbuf,
    bf16* __restrict__ states, float* __restrict__ chunksum) {
  __shared__ alignas(16) bf16 sC[64 * 72], sBm[64 * 72], sXT[64 * 72], sGT[64 * 72];
  __shared__ float sAc[8 * 64];
  __shared__ float sTot[8];
  __shared__ float sDt[64], sDec[64];
  int bc = blockIdx.x;
  int b = bc >> 4, c = bc & 15;
  int l0 = c * 64;
  int tid = threadIdx.x;
  int wave = tid >> 5;
  // async stage of Bm and C chunk tiles: 16 halves (32B) per thread each
  {
    int q = tid >> 2, seg = (tid & 3) * 16;
    const bf16* g = xbc + ((size_t)(b * L_ + l0 + q)) * CONVD_ + DI_ + seg;
    async_g2l_b128(g,            &sBm[q * 72 + seg]);
    async_g2l_b128(g + 8,        &sBm[q * 72 + seg + 8]);
    async_g2l_b128(g + DST_,     &sC [q * 72 + seg]);
    async_g2l_b128(g + DST_ + 8, &sC [q * 72 + seg + 8]);
  }
  if (tid < NH_) {
    int h = tid;
    float a = -__expf(alog[h]);
    float run = 0.f;
    for (int q = 0; q < 64; ++q) {
      run += dtb[((size_t)(b * L_ + l0 + q)) * NH_ + h] * a;
      sAc[h * 64 + q] = run;
    }
    sTot[h] = run;
    chunksum[(b * NH_ + h) * NCH + c] = run;
  }
  wait_async0();
  __syncthreads();
  // CB[q,s] = sum_n C[q,n] * Bm[s,n]
  for (int t = 0; t < 2; ++t) {
    int ti = wave * 2 + t;
    int tm = (ti >> 2) * 16, tn = (ti & 3) * 16;
    v8f acc = vzero8f();
    for (int k0 = 0; k0 < 64; k0 += 32)
      acc = wmma_bf16(ld_frag_a(sC, tm, k0, 72), ld_frag_b(sBm, tn, k0, 72), acc);
    int lane = tid & 31, r = lane & 15, hi = lane >> 4;
    for (int v = 0; v < 8; ++v)
      cbuf[(size_t)bc * 4096 + (tm + v + 8 * hi) * 64 + tn + r] = acc[v];
  }
  for (int h = 0; h < NH_; ++h) {
    __syncthreads();
    if (tid < 64) {
      sDt[tid]  = dtb[((size_t)(b * L_ + l0 + tid)) * NH_ + h];
      sDec[tid] = __expf(sTot[h] - sAc[h * 64 + tid]);
    }
    __syncthreads();
    for (int idx = tid; idx < 64 * 64; idx += 256) {
      int q = idx >> 6, j = idx & 63;
      size_t base = ((size_t)(b * L_ + l0 + q)) * CONVD_;
      sXT[j * 72 + q] = (bf16)((float)xbc[base + h * HD_ + j] * sDt[q]);   // Xdt^T
      sGT[j * 72 + q] = (bf16)((float)sBm[q * 72 + j] * sDec[q]);          // (Bm*dec)^T
    }
    __syncthreads();
    for (int t = 0; t < 2; ++t) {
      int ti = wave * 2 + t;
      int tm = (ti >> 2) * 16, tn = (ti & 3) * 16;
      v8f acc = vzero8f();
      for (int k0 = 0; k0 < 64; k0 += 32)
        acc = wmma_bf16(ld_frag_a(sXT, tm, k0, 72), ld_frag_b(sGT, tn, k0, 72), acc);
      int lane = tid & 31, r = lane & 15, hi = lane >> 4;
      for (int v = 0; v < 8; ++v)
        states[((size_t)bc * NH_ + h) * 4096 + (tm + v + 8 * hi) * 64 + tn + r] = (bf16)acc[v];
    }
  }
}

// ---------------------------------------------------------------------------
// Inter-chunk scan per (b, h): overwrite states_c with carried-in state N_c,
// N_0 = 0, N_{c+1} = N_c * exp(A_c) + states_c   (in place, 16 chunks)
// ---------------------------------------------------------------------------
__global__ __launch_bounds__(256) void scan_kernel(
    bf16* __restrict__ states, const float* __restrict__ chunksum) {
  int bh = blockIdx.x;          // b*NH + h
  int b = bh >> 3, h = bh & 7;
  int tid = threadIdx.x;        // each thread owns 16 of 4096 elements
  float N[16];
  for (int i = 0; i < 16; ++i) N[i] = 0.f;
  for (int c = 0; c < NCH; ++c) {
    size_t base = (((size_t)(b * NCH + c)) * NH_ + h) * 4096 + (size_t)tid * 16;
    float dec = __expf(chunksum[bh * NCH + c]);
    for (int i = 0; i < 16; ++i) {
      float s = (float)states[base + i];
      states[base + i] = (bf16)N[i];
      N[i] = N[i] * dec + s;
    }
  }
}

// ---------------------------------------------------------------------------
// SSD part 2, per (b, chunk, head):
// Y = (CB .* exp(Acum_q - Acum_s) .* tril) @ Xdt  +  (C .* exp(Acum_q)) @ prev^T
//   + Xs * Dskip[h]
// ---------------------------------------------------------------------------
__global__ __launch_bounds__(256) void ssd_part2(
    const bf16* __restrict__ xbc, const float* __restrict__ dtb,
    const float* __restrict__ alog, const float* __restrict__ cbuf,
    const bf16* __restrict__ prevst, const float* __restrict__ dskip,
    float* __restrict__ ybuf) {
  __shared__ alignas(16) bf16 sCr[64 * 72], sM[64 * 72], sXT[64 * 72], sCs[64 * 72], sN[64 * 72];
  __shared__ float sAc[8 * 64];
  __shared__ float sDt[64], sEA[64];
  int bc = blockIdx.x;
  int b = bc >> 4, c = bc & 15;
  int l0 = c * 64;
  int tid = threadIdx.x;
  int wave = tid >> 5;
  // async stage of C chunk tile
  {
    int q = tid >> 2, seg = (tid & 3) * 16;
    const bf16* g = xbc + ((size_t)(b * L_ + l0 + q)) * CONVD_ + DI_ + DST_ + seg;
    async_g2l_b128(g,     &sCr[q * 72 + seg]);
    async_g2l_b128(g + 8, &sCr[q * 72 + seg + 8]);
  }
  if (tid < NH_) {
    int h = tid;
    float a = -__expf(alog[h]);
    float run = 0.f;
    for (int q = 0; q < 64; ++q) {
      run += dtb[((size_t)(b * L_ + l0 + q)) * NH_ + h] * a;
      sAc[h * 64 + q] = run;
    }
  }
  wait_async0();
  for (int h = 0; h < NH_; ++h) {
    __syncthreads();
    if (tid < 64) {
      sDt[tid] = dtb[((size_t)(b * L_ + l0 + tid)) * NH_ + h];
      sEA[tid] = __expf(sAc[h * 64 + tid]);
    }
    __syncthreads();
    for (int idx = tid; idx < 64 * 64; idx += 256) {
      int q = idx >> 6, j = idx & 63;
      float cbv = cbuf[(size_t)bc * 4096 + q * 64 + j];
      float m = (j <= q) ? cbv * __expf(sAc[h * 64 + q] - sAc[h * 64 + j]) : 0.f;
      sM[q * 72 + j] = (bf16)m;
      size_t base = ((size_t)(b * L_ + l0 + q)) * CONVD_;
      sXT[j * 72 + q] = (bf16)((float)xbc[base + h * HD_ + j] * sDt[q]);   // Xdt^T
      sCs[q * 72 + j] = (bf16)((float)sCr[q * 72 + j] * sEA[q]);           // C * exp(Acum)
      sN [q * 72 + j] = prevst[((size_t)bc * NH_ + h) * 4096 + q * 64 + j];// prev[p][n]
    }
    __syncthreads();
    for (int t = 0; t < 2; ++t) {
      int ti = wave * 2 + t;
      int tm = (ti >> 2) * 16, tn = (ti & 3) * 16;
      v8f acc = vzero8f();
      for (int k0 = 0; k0 < 64; k0 += 32) {
        acc = wmma_bf16(ld_frag_a(sM,  tm, k0, 72), ld_frag_b(sXT, tn, k0, 72), acc); // Y_diag
        acc = wmma_bf16(ld_frag_a(sCs, tm, k0, 72), ld_frag_b(sN,  tn, k0, 72), acc); // Y_off
      }
      int lane = tid & 31, r = lane & 15, hi = lane >> 4;
      for (int v = 0; v < 8; ++v) {
        int q = tm + v + 8 * hi, p = tn + r;
        size_t row = (size_t)(b * L_ + l0 + q);
        float skip = (float)xbc[row * CONVD_ + h * HD_ + p] * dskip[h];
        ybuf[row * DI_ + h * HD_ + p] = acc[v] + skip;
      }
    }
  }
}

// ---------------------------------------------------------------------------
// y = y * silu(z); RMS-norm over DI; * gnorm_w; -> bf16
// ---------------------------------------------------------------------------
__global__ __launch_bounds__(256) void gated_rmsnorm(
    const float* __restrict__ ybuf, const float* __restrict__ zx,
    const float* __restrict__ gw, bf16* __restrict__ yb) {
  __shared__ float red[256];
  size_t row = blockIdx.x;
  int tid = threadIdx.x;
  float y0 = ybuf[row * DI_ + tid], y1 = ybuf[row * DI_ + tid + 256];
  float z0 = zx[row * DIP_ + tid],  z1 = zx[row * DIP_ + tid + 256];
  float g0 = y0 * siluf(z0);
  float g1 = y1 * siluf(z1);
  red[tid] = g0 * g0 + g1 * g1;
  __syncthreads();
  for (int s = 128; s > 0; s >>= 1) { if (tid < s) red[tid] += red[tid + s]; __syncthreads(); }
  float scale = rsqrtf(red[0] / (float)DI_ + 1e-5f);
  yb[row * DI_ + tid]       = (bf16)(g0 * scale * gw[tid]);
  yb[row * DI_ + tid + 256] = (bf16)(g1 * scale * gw[tid + 256]);
}

// ---------------------------------------------------------------------------
// h = layernorm(h + y_out) ; write f32 + bf16
// ---------------------------------------------------------------------------
__global__ __launch_bounds__(256) void residual_ln(
    float* __restrict__ hf, const float* __restrict__ yo,
    const float* __restrict__ g, const float* __restrict__ bb,
    bf16* __restrict__ hb) {
  __shared__ float red[256];
  size_t row = blockIdx.x;
  int tid = threadIdx.x;
  float v = hf[row * DM_ + tid] + yo[row * DM_ + tid];
  red[tid] = v; __syncthreads();
  for (int s = 128; s > 0; s >>= 1) { if (tid < s) red[tid] += red[tid + s]; __syncthreads(); }
  float m = red[0] / (float)DM_;
  __syncthreads();
  float d = v - m;
  red[tid] = d * d; __syncthreads();
  for (int s = 128; s > 0; s >>= 1) { if (tid < s) red[tid] += red[tid + s]; __syncthreads(); }
  float var = red[0] / (float)DM_;
  float o = d * rsqrtf(var + 1e-5f) * g[tid] + bb[tid];
  hf[row * DM_ + tid] = o;
  hb[row * DM_ + tid] = (bf16)o;
}

// ---------------------------------------------------------------------------
// De-normalize last PRED_ steps
// ---------------------------------------------------------------------------
__global__ void final_kernel(const float* __restrict__ ho, const float* __restrict__ stdb,
                             const float* __restrict__ meanb, float* __restrict__ out) {
  int idx = blockIdx.x * 256 + threadIdx.x;
  if (idx >= B_ * PRED_ * CO_) return;
  int c = idx % CO_;
  int t = (idx / CO_) % PRED_;
  int b = idx / (CO_ * PRED_);
  int l = L_ - PRED_ + t;
  out[idx] = ho[((size_t)(b * L_ + l)) * CO_ + c] * stdb[b * ENC + c] + meanb[b * ENC + c];
}

// ---------------------------------------------------------------------------
// Launch
// ---------------------------------------------------------------------------
extern "C" void kernel_launch(void* const* d_in, const int* in_sizes, int n_in,
                              void* d_out, int out_size, void* d_ws, size_t ws_size,
                              hipStream_t stream) {
  const float* x_enc   = (const float*)d_in[0];
  const float* x_mark  = (const float*)d_in[1];
  // d_in[2], d_in[3] (x_dec, x_mark_dec) unused by the reference path
  const float* token_w = (const float*)d_in[4];
  const float* temb_w  = (const float*)d_in[5];
  const float* ln_g    = (const float*)d_in[6];
  const float* ln_b    = (const float*)d_in[7];
  const float* in_w    = (const float*)d_in[8];
  const float* conv_w  = (const float*)d_in[9];
  const float* conv_b  = (const float*)d_in[10];
  const float* dt_bias = (const float*)d_in[11];
  const float* A_log   = (const float*)d_in[12];
  const float* Dskip   = (const float*)d_in[13];
  const float* gnorm_w = (const float*)d_in[14];
  const float* out_w   = (const float*)d_in[15];
  const float* head_w  = (const float*)d_in[16];
  float* out = (float*)d_out;
  (void)in_sizes; (void)n_in; (void)out_size; (void)ws_size;

  char* ws = (char*)d_ws;
  size_t off = 0;
  auto alloc = [&](size_t bytes) { size_t o = off; off += (bytes + 255) & ~(size_t)255; return o; };
  float* meanb = (float*)(ws + alloc((size_t)B_ * ENC * 4));
  float* stdb  = (float*)(ws + alloc((size_t)B_ * ENC * 4));
  float* hf    = (float*)(ws + alloc((size_t)TOK * DM_ * 4));
  bf16*  hb    = (bf16*) (ws + alloc((size_t)TOK * DM_ * 2));
  float* zx    = (float*)(ws + alloc((size_t)TOK * DIP_ * 4));
  bf16*  xbc   = (bf16*) (ws + alloc((size_t)TOK * CONVD_ * 2));
  float* dtb   = (float*)(ws + alloc((size_t)TOK * NH_ * 4));
  float* cbuf  = (float*)(ws + alloc((size_t)B_ * NCH * 4096 * 4));
  float* csum  = (float*)(ws + alloc((size_t)B_ * NH_ * NCH * 4));
  bf16*  stbuf = (bf16*) (ws + alloc((size_t)B_ * NCH * NH_ * 4096 * 2));
  float* ybuf  = (float*)(ws + alloc((size_t)TOK * DI_ * 4));
  bf16*  yb    = (bf16*) (ws + alloc((size_t)TOK * DI_ * 2));
  float* tmpb  = (float*)(ws + alloc((size_t)TOK * DM_ * 4));
  float* hout  = (float*)(ws + alloc((size_t)TOK * CO_ * 4));
  bf16*  w_in  = (bf16*) (ws + alloc((size_t)NL_ * DIP_ * DM_ * 2));
  bf16*  w_out = (bf16*) (ws + alloc((size_t)NL_ * DM_ * DI_ * 2));
  bf16*  w_head= (bf16*) (ws + alloc((size_t)CO_ * DM_ * 2));
  (void)alloc((size_t)64 * DM_ * 2);   // tail pad: unused, GEMM W loads are guarded at the N edge

  stats_kernel<<<B_ * ENC, 256, 0, stream>>>(x_enc, meanb, stdb);
  {
    int n = NL_ * DIP_ * DM_;
    cvt_bf16<<<(n + 255) / 256, 256, 0, stream>>>(in_w, w_in, n);
    n = NL_ * DM_ * DI_;
    cvt_bf16<<<(n + 255) / 256, 256, 0, stream>>>(out_w, w_out, n);
    n = CO_ * DM_;
    cvt_bf16<<<(n + 255) / 256, 256, 0, stream>>>(head_w, w_head, n);
  }
  embed_kernel<<<TOK, 256, 0, stream>>>(x_enc, x_mark, token_w, temb_w, meanb, stdb, hf, hb);

  for (int layer = 0; layer < NL_; ++layer) {
    gemm_bf16_nt<<<dim3(TOK / 128, (DIP_ + 63) / 64), 256, 0, stream>>>(
        hb, w_in + (size_t)layer * DIP_ * DM_, zx, TOK, DIP_, DM_);
    conv_silu_kernel<<<TOK, 256, 0, stream>>>(
        zx, conv_w + (size_t)layer * CONVD_ * DC_, conv_b + (size_t)layer * CONVD_,
        dt_bias + layer * NH_, xbc, dtb);
    ssd_part1<<<B_ * NCH, 256, 0, stream>>>(xbc, dtb, A_log + layer * NH_, cbuf, stbuf, csum);
    scan_kernel<<<B_ * NH_, 256, 0, stream>>>(stbuf, csum);
    ssd_part2<<<B_ * NCH, 256, 0, stream>>>(xbc, dtb, A_log + layer * NH_, cbuf, stbuf,
                                            Dskip + layer * NH_, ybuf);
    gated_rmsnorm<<<TOK, 256, 0, stream>>>(ybuf, zx, gnorm_w + layer * DI_, yb);
    gemm_bf16_nt<<<dim3(TOK / 128, DM_ / 64), 256, 0, stream>>>(
        yb, w_out + (size_t)layer * DM_ * DI_, tmpb, TOK, DM_, DI_);
    residual_ln<<<TOK, 256, 0, stream>>>(hf, tmpb, ln_g + layer * DM_, ln_b + layer * DM_, hb);
  }

  gemm_bf16_nt<<<dim3(TOK / 128, 1), 256, 0, stream>>>(hb, w_head, hout, TOK, CO_, DM_);
  final_kernel<<<(B_ * PRED_ * CO_ + 255) / 256, 256, 0, stream>>>(hout, stdb, meanb, out);
}
